// FastformerAttention_73358041415925
// MI455X (gfx1250) — matrix operand order repaired and checked by reading, not compiled, not run on hardware
//
#include <hip/hip_runtime.h>
#include <hip/hip_bf16.h>
#include <math.h>

// ---------------------------------------------------------------------------
// Fastformer (additive attention) for MI455X / gfx1250, wave32 + WMMA bf16.
//   B=32, N=4096, DIM=512, DD=512
// out = (gk (.) v) @ Wr + q  ==  x @ (Wv . diag(gk) . Wr) + q
// ---------------------------------------------------------------------------

#define FB   32
#define FN   4096
#define FDIM 512
#define FDD  512
#define FSCALE 0.04419417382415922f   // 512^-0.5
#define TB 256

typedef __attribute__((ext_vector_type(16))) __bf16 v16bf;
typedef __attribute__((ext_vector_type(8)))  __bf16 v8bf;
typedef __attribute__((ext_vector_type(8)))  float  v8f;

// ---------------------------------------------------------------------------
// K0: convert weights to bf16 in WMMA-B-friendly [n][k] layout, zero pools.
// ---------------------------------------------------------------------------
__global__ __launch_bounds__(TB) void ff_prep_kernel(
    const float* __restrict__ Wq, const float* __restrict__ Wk,
    const float* __restrict__ Wv, const float* __restrict__ Wr,
    __bf16* __restrict__ WqT, __bf16* __restrict__ WkT,
    __bf16* __restrict__ Wvb, __bf16* __restrict__ WrT,
    float* __restrict__ gq, float* __restrict__ gk)
{
    int tid = blockIdx.x * TB + threadIdx.x;
    if (tid < FB * FDD) { gq[tid] = 0.0f; gk[tid] = 0.0f; }
    if (tid < FDIM * FDD) {
        int r = tid / FDD;          // input row of W  (k index)
        int c = tid - r * FDD;      // output column   (n index)
        WqT[(size_t)c * FDIM + r] = (__bf16)Wq[tid];   // [n][k]
        WkT[(size_t)c * FDIM + r] = (__bf16)Wk[tid];   // [n][k]
        Wvb[tid]                  = (__bf16)Wv[tid];   // [i][d] (= B^T for weff)
        WrT[(size_t)c * FDD  + r] = (__bf16)Wr[tid];   // WrT[e][d]
    }
}

// ---------------------------------------------------------------------------
// K1/K2: projection GEMM (32 rows x 512 cols per WG, M-block=2) fused with
// per-row softmax over the feature dim and pooled accumulation.
//   mode 0: y = x@Wq ; pool += y * softmax(y*alpha*S) ; store y (q)
//   mode 1: y = gq (.) (x@Wk) ; pool += y * softmax(y*beta*S)
// grid: B*N/32 = 4096 WGs x 256 threads (8 waves, each owns 64 columns).
// ---------------------------------------------------------------------------
__global__ __launch_bounds__(TB) void ff_proj_pool_kernel(
    const float* __restrict__ x, const __bf16* __restrict__ WT,
    const float* __restrict__ coef, const float* __restrict__ g_in,
    float* __restrict__ q_out, float* __restrict__ pool_out, int mode)
{
    __shared__ __bf16 lds_x[32 * FDIM];   // 32 KB  bf16 x tile (2 row-tiles)
    __shared__ float  lds_t[32 * FDD];    // 64 KB  fp32 result tile
    __shared__ float  lds_pool[FDD];      //  2 KB  per-WG pooled partial
    __shared__ float  lds_g[FDD];         //  2 KB  broadcast global_query
    __shared__ float  lds_c[FDD];         //  2 KB  alpha / beta

    const int wg  = blockIdx.x;
    const int b   = wg >> 7;              // 128 row tiles (of 32) per batch
    const int m0  = (wg & 127) << 5;
    const int tid = threadIdx.x;
    const float* xrow = x + ((size_t)b * FN + m0) * FDIM;

    // stage x tile as bf16 + coefficients; prefetch next tile into caches
    for (int i = tid; i < 32 * FDIM; i += TB) lds_x[i] = (__bf16)xrow[i];
    for (int i = tid; i < FDD; i += TB) {
        lds_c[i]    = coef[i];
        lds_pool[i] = 0.0f;
        lds_g[i]    = mode ? g_in[(size_t)b * FDD + i] : 1.0f;
    }
    if (wg + 1 < (int)gridDim.x)
        __builtin_prefetch(xrow + (size_t)32 * FDIM + threadIdx.x * 64, 0, 0);
    __syncthreads();

    // ---- WMMA GEMM: 2 M-tiles x 4 N-tiles per wave; B fragments reused ----
    const int wave  = tid >> 5;
    const int lane  = tid & 31;
    const int mrow  = lane & 15;
    const int kA    = (lane < 16) ? 0 : 16;   // A-fragment K sub-offset
    const int kB    = (lane < 16) ? 0 : 16;   // B-fragment K sub-offset
    const int cbase = wave * 64;

    v8f acc[2][4] = {};
    for (int ks = 0; ks < FDIM; ks += 32) {
        v16bf afrag[2];
#pragma unroll
        for (int m = 0; m < 2; ++m) {
            const int arow = m * 16 + mrow;
            v8bf lo = *(const v8bf*)&lds_x[arow * FDIM + ks + kA];
            v8bf hi = *(const v8bf*)&lds_x[arow * FDIM + ks + kA + 8];
            afrag[m] = __builtin_shufflevector(lo, hi,
                0,1,2,3,4,5,6,7,8,9,10,11,12,13,14,15);
        }
#pragma unroll
        for (int ct = 0; ct < 4; ++ct) {
            const int col = cbase + ct * 16 + (lane & 15);
            v16bf bfrag = *(const v16bf*)&WT[(size_t)col * FDIM + ks + kB];
#pragma unroll
            for (int m = 0; m < 2; ++m)
                acc[m][ct] = __builtin_amdgcn_wmma_f32_16x16x32_bf16(
                    false, afrag[m], false, bfrag, (short)0, acc[m][ct],
                    false, false);
        }
    }
#pragma unroll
    for (int m = 0; m < 2; ++m)
#pragma unroll
        for (int ct = 0; ct < 4; ++ct) {
            const int col   = cbase + ct * 16 + (lane & 15);
            const int rbase = m * 16 + ((lane < 16) ? 0 : 8);
#pragma unroll
            for (int r = 0; r < 8; ++r)
                lds_t[(rbase + r) * FDD + col] = acc[m][ct][r];
        }
    __syncthreads();

    // ---- fused per-row softmax over feature dim + pooled accumulation ----
    // 16 threads per row; two 16-row halves. xor masks 1,2,4,8 stay inside
    // each 16-lane half of the wave.
    const int sub = tid & 15;
#pragma unroll
    for (int half = 0; half < 2; ++half) {
        const int row = (tid >> 4) + (half << 4);
        float vloc[32], lg[32];
        float mx = -3.402823466e38f;
#pragma unroll
        for (int j = 0; j < 32; ++j) {
            const int col = sub + (j << 4);
            float v = lds_t[row * FDD + col] * lds_g[col];   // mode0: *1.0
            vloc[j] = v;
            float l = v * lds_c[col] * FSCALE;
            lg[j] = l;
            mx = fmaxf(mx, l);
        }
#pragma unroll
        for (int off = 8; off; off >>= 1) mx = fmaxf(mx, __shfl_xor(mx, off, 32));
        float se = 0.0f;
#pragma unroll
        for (int j = 0; j < 32; ++j) { float e = __expf(lg[j] - mx); lg[j] = e; se += e; }
#pragma unroll
        for (int off = 8; off; off >>= 1) se += __shfl_xor(se, off, 32);
        const float inv = 1.0f / se;
#pragma unroll
        for (int j = 0; j < 32; ++j)
            atomicAdd(&lds_pool[sub + (j << 4)], vloc[j] * lg[j] * inv);
    }
    __syncthreads();

    for (int i = tid; i < FDD; i += TB)
        atomicAdd(&pool_out[(size_t)b * FDD + i], lds_pool[i]);

    if (mode == 0) {   // persist q (fp32) for the residual in the last pass
        float* qdst = q_out + ((size_t)b * FN + m0) * FDD;
        for (int i = tid; i < 32 * FDD; i += TB) qdst[i] = lds_t[i];
    }
}

// ---------------------------------------------------------------------------
// K3: per-batch W_effT[e][i] = sum_d WrT[e][d]*gk[d]*WvT[d][i]  (bf16 WMMA)
// Output stored directly in [n=e][k=i] B-fragment layout for the final GEMM.
// grid: 32 tiles/batch * 32 batches = 1024 WGs x 256 threads.
// ---------------------------------------------------------------------------
__global__ __launch_bounds__(TB) void ff_weff_kernel(
    const __bf16* __restrict__ WrT, const __bf16* __restrict__ Wvb,
    const float* __restrict__ gk, __bf16* __restrict__ WeffT)
{
    __shared__ __bf16 lds_a[16 * FDD];   // gk-scaled WrT rows (16 KB)

    const int wg  = blockIdx.x;
    const int b   = wg >> 5;
    const int e0  = (wg & 31) << 4;
    const int tid = threadIdx.x;

    const __bf16* arow = WrT + (size_t)e0 * FDD;
    const float*  g    = gk + (size_t)b * FDD;
    for (int i = tid; i < 16 * FDD; i += TB) {
        const int d = i & (FDD - 1);
        lds_a[i] = (__bf16)((float)arow[i] * g[d]);
    }
    __syncthreads();

    const int wave  = tid >> 5;
    const int lane  = tid & 31;
    const int mrow  = lane & 15;
    const int kA    = (lane < 16) ? 0 : 16;
    const int kB    = (lane < 16) ? 0 : 16;
    const int cbase = wave * 64;

    v8f acc[4] = {};
    for (int ks = 0; ks < FDD; ks += 32) {
        v8bf lo = *(const v8bf*)&lds_a[mrow * FDD + ks + kA];
        v8bf hi = *(const v8bf*)&lds_a[mrow * FDD + ks + kA + 8];
        v16bf afrag = __builtin_shufflevector(lo, hi,
            0,1,2,3,4,5,6,7,8,9,10,11,12,13,14,15);
#pragma unroll
        for (int ct = 0; ct < 4; ++ct) {
            const int col = cbase + ct * 16 + (lane & 15);   // i index
            v16bf bfrag = *(const v16bf*)&Wvb[(size_t)col * FDD + ks + kB];
            acc[ct] = __builtin_amdgcn_wmma_f32_16x16x32_bf16(
                false, afrag, false, bfrag, (short)0, acc[ct], false, false);
        }
    }
    __bf16* dst = WeffT + (size_t)b * FDD * FDIM;
#pragma unroll
    for (int ct = 0; ct < 4; ++ct) {
        const int col   = cbase + ct * 16 + (lane & 15);
        const int rbase = (lane < 16) ? 0 : 8;
#pragma unroll
        for (int r = 0; r < 8; ++r)
            dst[(size_t)(e0 + rbase + r) * FDIM + col] = (__bf16)acc[ct][r];
    }
}

// ---------------------------------------------------------------------------
// K4: out = x @ W_eff[b] + q   (bf16 WMMA, M-block=4: 64 rows per WG)
// grid: B*N/64 = 2048 WGs x 256 threads. Each B fragment feeds 4 wmmas.
// ---------------------------------------------------------------------------
__global__ __launch_bounds__(TB) void ff_out_kernel(
    const float* __restrict__ x, const __bf16* __restrict__ WeffT,
    const float* __restrict__ qbuf, float* __restrict__ out)
{
    __shared__ __bf16 lds_x[64 * FDIM];   // 64 KB

    const int wg  = blockIdx.x;
    const int b   = wg >> 6;              // 64 row tiles (of 64) per batch
    const int m0  = (wg & 63) << 6;
    const int tid = threadIdx.x;
    const float* xrow = x + ((size_t)b * FN + m0) * FDIM;

    for (int i = tid; i < 64 * FDIM; i += TB) lds_x[i] = (__bf16)xrow[i];
    if (wg + 1 < (int)gridDim.x)
        __builtin_prefetch(xrow + (size_t)64 * FDIM + threadIdx.x * 64, 0, 0);
    __syncthreads();

    const int wave  = tid >> 5;
    const int lane  = tid & 31;
    const int mrow  = lane & 15;
    const int kA    = (lane < 16) ? 0 : 16;
    const int kB    = (lane < 16) ? 0 : 16;
    const int cbase = wave * 64;
    const __bf16* WB = WeffT + (size_t)b * FDD * FDIM;

    v8f acc[4][4] = {};
    for (int ks = 0; ks < FDIM; ks += 32) {
        v16bf afrag[4];
#pragma unroll
        for (int m = 0; m < 4; ++m) {
            const int arow = m * 16 + mrow;
            v8bf lo = *(const v8bf*)&lds_x[arow * FDIM + ks + kA];
            v8bf hi = *(const v8bf*)&lds_x[arow * FDIM + ks + kA + 8];
            afrag[m] = __builtin_shufflevector(lo, hi,
                0,1,2,3,4,5,6,7,8,9,10,11,12,13,14,15);
        }
#pragma unroll
        for (int ct = 0; ct < 4; ++ct) {
            const int col = cbase + ct * 16 + (lane & 15);
            v16bf bfrag = *(const v16bf*)&WB[(size_t)col * FDIM + ks + kB];
#pragma unroll
            for (int m = 0; m < 4; ++m)
                acc[m][ct] = __builtin_amdgcn_wmma_f32_16x16x32_bf16(
                    false, afrag[m], false, bfrag, (short)0, acc[m][ct],
                    false, false);
        }
    }
    const size_t rowbase = (size_t)b * FN + m0;
#pragma unroll
    for (int m = 0; m < 4; ++m)
#pragma unroll
        for (int ct = 0; ct < 4; ++ct) {
            const int col   = cbase + ct * 16 + (lane & 15);
            const int rbase = m * 16 + ((lane < 16) ? 0 : 8);
#pragma unroll
            for (int r = 0; r < 8; ++r) {
                const size_t idx = (rowbase + rbase + r) * FDD + col;
                out[idx] = acc[m][ct][r] + qbuf[idx];
            }
        }
}

// ---------------------------------------------------------------------------
// launch
// ---------------------------------------------------------------------------
extern "C" void kernel_launch(void* const* d_in, const int* in_sizes, int n_in,
                              void* d_out, int out_size, void* d_ws, size_t ws_size,
                              hipStream_t stream)
{
    const float* x     = (const float*)d_in[0];
    const float* Wq    = (const float*)d_in[1];
    const float* Wk    = (const float*)d_in[2];
    const float* Wv    = (const float*)d_in[3];
    const float* Wr    = (const float*)d_in[4];
    const float* alpha = (const float*)d_in[5];
    const float* beta  = (const float*)d_in[6];
    float* out = (float*)d_out;

    char* ws = (char*)d_ws;
    size_t off = 0;
    float*  qbuf  = (float*)(ws + off); off += (size_t)FB * FN * FDD * sizeof(float);
    float*  gq    = (float*)(ws + off); off += (size_t)FB * FDD * sizeof(float);
    float*  gk    = (float*)(ws + off); off += (size_t)FB * FDD * sizeof(float);
    __bf16* WqT   = (__bf16*)(ws + off); off += (size_t)FDIM * FDD * 2;
    __bf16* WkT   = (__bf16*)(ws + off); off += (size_t)FDIM * FDD * 2;
    __bf16* Wvb   = (__bf16*)(ws + off); off += (size_t)FDIM * FDD * 2;
    __bf16* WrT   = (__bf16*)(ws + off); off += (size_t)FDD * FDD * 2;
    __bf16* WeffT = (__bf16*)(ws + off); off += (size_t)FB * FDD * FDIM * 2;

    dim3 blk(TB);

    // K0: weight convert/transpose + zero pooled accumulators
    ff_prep_kernel<<<dim3((FDIM * FDD + TB - 1) / TB), blk, 0, stream>>>(
        Wq, Wk, Wv, Wr, WqT, WkT, Wvb, WrT, gq, gk);

    // K1: q = x@Wq, global_query pooling (alpha)
    ff_proj_pool_kernel<<<dim3(FB * FN / 32), blk, 0, stream>>>(
        x, WqT, alpha, gq /*unused*/, qbuf, gq, 0);

    // K2: k = x@Wk, p = gq(.)k, global_key pooling (beta)
    ff_proj_pool_kernel<<<dim3(FB * FN / 32), blk, 0, stream>>>(
        x, WkT, beta, gq, qbuf /*unused*/, gk, 1);

    // K3: per-batch W_effT = WrT . diag(gk) . WvT
    ff_weff_kernel<<<dim3(FB * 32), blk, 0, stream>>>(WrT, Wvb, gk, WeffT);

    // K4: out = x @ W_eff + q
    ff_out_kernel<<<dim3(FB * FN / 64), blk, 0, stream>>>(x, WeffT, qbuf, out);
}
